// PPFLocalGlobalNet_317827580542
// MI455X (gfx1250) — matrix operand order repaired and checked by reading, not compiled, not run on hardware
//
#include <hip/hip_runtime.h>
#include <math.h>

typedef __attribute__((ext_vector_type(16))) _Float16 v16h;
typedef __attribute__((ext_vector_type(8)))  _Float16 v8h;
typedef __attribute__((ext_vector_type(8)))  float    v8f;

#define NPTS     4096
#define BATCH    2
#define NSAMP    64
#define RAD2     0.09f
#define NC_PRE   524288L   /* B*S*N */
#define NC_POST  8192L     /* B*N   */
#define SN       262144L   /* S*N   */

// ---------------------------------------------------------------- utilities
__global__ void zero_f32(float* __restrict__ p, int n) {
  int t = blockIdx.x * blockDim.x + threadIdx.x;
  if (t < n) p[t] = 0.f;
}

// Weights (M,K) f32 -> (M,K32) f16, zero-padded in K. grid=(M), block=(K32).
__global__ void cvt_pad(const float* __restrict__ src, _Float16* __restrict__ dst,
                        int K, int K32) {
  const int m = blockIdx.x, k = threadIdx.x;
  dst[m * K32 + k] = (k < K) ? (_Float16)src[m * K + k] : (_Float16)0.f;
}

// ---------------------------------------------------------------- ball grouping
// One wave per point. Scan j ascending, collect first 64 with d2<r^2 via
// ballot + prefix-popcount compaction; fill with the first hit (self always hits).
__global__ void ball_group_kernel(const float* __restrict__ xyz, int* __restrict__ idx) {
  const int lane = threadIdx.x & 31;
  const int p    = blockIdx.x * (blockDim.x >> 5) + (threadIdx.x >> 5); // b*N+i
  const int b    = p >> 12;
  const float px = xyz[(long)p * 3 + 0];
  const float py = xyz[(long)p * 3 + 1];
  const float pz = xyz[(long)p * 3 + 2];
  int* __restrict__ out = idx + (long)p * NSAMP;
  int count  = 0;
  int firstJ = p & 4095;
  for (int j0 = 0; j0 < NPTS && count < NSAMP; j0 += 32) {
    const int  j = j0 + lane;
    const long q = ((long)b * NPTS + j) * 3;
    const float dx = xyz[q + 0] - px;
    const float dy = xyz[q + 1] - py;
    const float dz = xyz[q + 2] - pz;
    const bool w = (dx * dx + dy * dy + dz * dz) < RAD2;
    const unsigned mask = __builtin_amdgcn_ballot_w32(w);
    if (count == 0 && mask) firstJ = j0 + (__ffs(mask) - 1);
    const int prefix = __popc(mask & ((1u << lane) - 1u));
    const int slot   = count + prefix;
    if (w && slot < NSAMP) out[slot] = j;
    count += __popc(mask);
  }
  if (count > NSAMP) count = NSAMP;
  for (int s = count + lane; s < NSAMP; s += 32) out[s] = firstJ;
}

// ---------------------------------------------------------------- PPF features
__device__ __forceinline__ float ppf_angle(float ax, float ay, float az,
                                           float bx, float by, float bz) {
  const float cx = ay * bz - az * by;
  const float cy = az * bx - ax * bz;
  const float cz = ax * by - ay * bx;
  const float cn = sqrtf(cx * cx + cy * cy + cz * cz + 1e-12f);
  return atan2f(cn, ax * bx + ay * by + az * bz);
}

// Writes column-major X[col][32]: 4 features + 28 zero-pad (K padded to 32).
__global__ void ppf_kernel(const float* __restrict__ xyz, const float* __restrict__ nrm,
                           const int* __restrict__ idx, _Float16* __restrict__ X) {
  const long col = (long)blockIdx.x * blockDim.x + threadIdx.x; // (b,s,n) flatten
  const int  n = (int)(col & 4095);
  const int  s = (int)((col >> 12) & 63);
  const int  b = (int)(col >> 18);
  const long bi = (long)b * NPTS + n;
  const int  j  = idx[bi * NSAMP + s];
  const long qi = bi * 3;
  const long qj = ((long)b * NPTS + j) * 3;
  const float dx = xyz[qj + 0] - xyz[qi + 0];
  const float dy = xyz[qj + 1] - xyz[qi + 1];
  const float dz = xyz[qj + 2] - xyz[qi + 2];
  const float nix = nrm[qi + 0], niy = nrm[qi + 1], niz = nrm[qi + 2];
  const float njx = nrm[qj + 0], njy = nrm[qj + 1], njz = nrm[qj + 2];
  const float f0 = ppf_angle(nix, niy, niz, dx, dy, dz);
  const float f1 = ppf_angle(njx, njy, njz, dx, dy, dz);
  const float f2 = ppf_angle(nix, niy, niz, njx, njy, njz);
  const float f3 = sqrtf(dx * dx + dy * dy + dz * dz + 1e-12f);
  v8h h0 = {(_Float16)f0, (_Float16)f1, (_Float16)f2, (_Float16)f3,
            (_Float16)0.f, (_Float16)0.f, (_Float16)0.f, (_Float16)0.f};
  v8h z  = {};
  v8h* o = (v8h*)(X + col * 32);
  o[0] = h0; o[1] = z; o[2] = z; o[3] = z;
}

// ---------------------------------------------------------------- WMMA GEMM
// Column-major activations: X[col][K32], Y[col][M].
// Y[col][m] = sum_k W[m][k]*X[col][k] + bias[m]; all fragment traffic is b128.
// Each wave: 16(M) x 64(N) via 4 accumulators reusing one A fragment.
// Fused GroupNorm statistics: LDS atomic reduction -> global atomics.
// grid: x = NC/512, y = M/16. block = 256 (8 waves).
__global__ void gemm_wmma(const _Float16* __restrict__ X, const _Float16* __restrict__ W,
                          const float* __restrict__ bias, _Float16* __restrict__ Y,
                          float* __restrict__ stats, int M, int K32, long NC, int cbShift) {
  const int wave = threadIdx.x >> 5;
  const int lane = threadIdx.x & 31;
  const int row  = lane & 15;
  const int hi   = lane >> 4;
  const int m0   = blockIdx.y * 16;
  const long colBase = ((long)blockIdx.x * 8 + wave) * 64;

  v8f acc[4] = {{}, {}, {}, {}};
  const _Float16* __restrict__ wp = W + (long)(m0 + row) * K32;
  for (int k0 = 0; k0 < K32; k0 += 32) {
    // A fragment 16x32: lane(hi) holds k 8*hi..8*hi+7 and 16+8*hi..16+8*hi+7; two contiguous b128 loads.
    const v8h alo = *(const v8h*)(wp + k0 + 8 * hi);
    const v8h ahi = *(const v8h*)(wp + k0 + 16 + 8 * hi);
    const v16h a = __builtin_shufflevector(alo, ahi, 0, 1, 2, 3, 4, 5, 6, 7,
                                           8, 9, 10, 11, 12, 13, 14, 15);
#pragma unroll
    for (int j = 0; j < 4; ++j) {
      // B fragment 32x16 from column-major X: K-contiguous -> two b128 loads.
      const _Float16* __restrict__ xp =
          X + (colBase + j * 16 + row) * K32 + k0 + 8 * hi;
      const v8h blo = *(const v8h*)xp;
      const v8h bhi = *(const v8h*)(xp + 16);
      const v16h bf = __builtin_shufflevector(blo, bhi, 0, 1, 2, 3, 4, 5, 6, 7,
                                              8, 9, 10, 11, 12, 13, 14, 15);
      acc[j] = __builtin_amdgcn_wmma_f32_16x16x32_f16(false, a, false, bf,
                                                      (short)0, acc[j], false, false);
    }
  }

  __shared__ float lsum[16];
  __shared__ float lsq[16];
  if (stats) {
    if (threadIdx.x < 16) { lsum[threadIdx.x] = 0.f; lsq[threadIdx.x] = 0.f; }
    __syncthreads();
  }

  const int chPerG = M >> 3;
  const int mBase  = m0 + 8 * hi;       // D layout: VGPR r -> M = r + 8*hi
#pragma unroll
  for (int j = 0; j < 4; ++j) {
    const long col  = colBase + j * 16 + row;
    const int  bidx = (int)(col >> cbShift);
    v8h o;
#pragma unroll
    for (int r = 0; r < 8; ++r) {
      const float v = acc[j][r] + bias[mBase + r];
      o[r] = (_Float16)v;
      if (stats) {
        const int bg = bidx * 8 + (mBase + r) / chPerG;
        atomicAdd(&lsum[bg], v);
        atomicAdd(&lsq[bg], v * v);
      }
    }
    *(v8h*)(Y + col * M + mBase) = o;   // one b128 store per tile
  }

  if (stats) {
    __syncthreads();
    if (threadIdx.x < 16) {
      atomicAdd(&stats[threadIdx.x],      lsum[threadIdx.x]);
      atomicAdd(&stats[16 + threadIdx.x], lsq[threadIdx.x]);
    }
  }
}

// ---------------------------------------------------------------- GN finalize / apply
__global__ void finalize_stats(const float* __restrict__ stats, float* __restrict__ mv,
                               float invCnt) {
  const int t = threadIdx.x; // 16 threads
  const float mean = stats[t] * invCnt;
  const float var  = stats[16 + t] * invCnt - mean * mean;
  mv[t]      = mean;
  mv[16 + t] = rsqrtf(var + 1e-5f);
}

// Column-major GN + ReLU: one thread per column, contiguous channel vector.
__global__ void gn_relu_col(const _Float16* __restrict__ raw, _Float16* __restrict__ act,
                            const float* __restrict__ mv, const float* __restrict__ gamma,
                            const float* __restrict__ beta, int M, int cbShift) {
  const long col = (long)blockIdx.x * blockDim.x + threadIdx.x;
  const int  b   = (int)(col >> cbShift);
  const int  chPerG = M >> 3;
  const long base = col * M;
  for (int m = 0; m < M; ++m) {
    const int bg = b * 8 + m / chPerG;
    float v = (float)raw[base + m];
    v = (v - mv[bg]) * mv[16 + bg] * gamma[m] + beta[m];
    v = fmaxf(v, 0.f);
    act[base + m] = (_Float16)v;
  }
}

// ---------------------------------------------------------------- max over samples
// Block per output point (b*N+n), thread per channel; coalesced channel reads.
__global__ void maxpool_col(const _Float16* __restrict__ act, _Float16* __restrict__ pooled) {
  const int bn = blockIdx.x;      // 0..8191
  const int c  = threadIdx.x;     // 0..191
  const int b  = bn >> 12;
  const int n  = bn & 4095;
  const long colBase = (long)b * SN + n;
  float m = -1e30f;
  for (int s = 0; s < NSAMP; ++s)
    m = fmaxf(m, (float)act[(colBase + (long)s * NPTS) * 192 + c]);
  pooled[(long)bn * 192 + c] = (_Float16)m;
}

// ---------------------------------------------------------------- final L2 norm
__global__ void final_norm(const _Float16* __restrict__ Y, float* __restrict__ out) {
  const int p = blockIdx.x * blockDim.x + threadIdx.x; // b*N+n
  const _Float16* __restrict__ y = Y + (long)p * 96;
  float ss = 0.f;
  for (int c = 0; c < 96; ++c) {
    const float v = (float)y[c];
    ss += v * v;
  }
  const float inv = 1.f / sqrtf(ss);
  for (int c = 0; c < 96; ++c) out[(long)p * 96 + c] = (float)y[c] * inv;
}

// ---------------------------------------------------------------- host orchestration
extern "C" void kernel_launch(void* const* d_in, const int* in_sizes, int n_in,
                              void* d_out, int out_size, void* d_ws, size_t ws_size,
                              hipStream_t stream) {
  const float* xyz = (const float*)d_in[0];
  const float* nrm = (const float*)d_in[1];
  // params dict insertion order: pre{0..2}: W,b,g,bt ; post0,post1: W,b,g,bt ; post2: W,b
  const float* preW[3]  = {(const float*)d_in[2],  (const float*)d_in[6],  (const float*)d_in[10]};
  const float* preB[3]  = {(const float*)d_in[3],  (const float*)d_in[7],  (const float*)d_in[11]};
  const float* preG[3]  = {(const float*)d_in[4],  (const float*)d_in[8],  (const float*)d_in[12]};
  const float* preBt[3] = {(const float*)d_in[5],  (const float*)d_in[9],  (const float*)d_in[13]};
  const float* postW[3]  = {(const float*)d_in[14], (const float*)d_in[18], (const float*)d_in[22]};
  const float* postB[3]  = {(const float*)d_in[15], (const float*)d_in[19], (const float*)d_in[23]};
  const float* postG[2]  = {(const float*)d_in[16], (const float*)d_in[20]};
  const float* postBt[2] = {(const float*)d_in[17], (const float*)d_in[21]};

  // ---- carve workspace (256B aligned)
  size_t off = 0;
  char* base = (char*)d_ws;
  auto carve = [&](size_t bytes) -> void* {
    void* p = base + off;
    off += (bytes + 255) & ~(size_t)255;
    return p;
  };
  int*       idxBuf = (int*)carve((size_t)BATCH * NPTS * NSAMP * 4);
  _Float16*  ppfh   = (_Float16*)carve((size_t)NC_PRE * 32 * 2);   // [col][32], K-padded
  _Float16*  RA     = (_Float16*)carve((size_t)NC_PRE * 192 * 2);  // [col][M] ping
  _Float16*  RB     = (_Float16*)carve((size_t)NC_PRE * 192 * 2);  // [col][M] pong
  _Float16*  pooled = (_Float16*)carve((size_t)NC_POST * 192 * 2); // [bn][192]
  _Float16*  SA     = (_Float16*)carve((size_t)NC_POST * 192 * 2);
  _Float16*  SB     = (_Float16*)carve((size_t)NC_POST * 192 * 2);
  // padded f16 weights: (M, K32)
  const int wM[6]   = {96, 96, 192, 192, 96, 96};
  const int wK[6]   = {4, 96, 96, 192, 192, 96};
  const int wK32[6] = {32, 96, 96, 192, 192, 96};
  _Float16* Wh[6];
  for (int l = 0; l < 6; ++l) Wh[l] = (_Float16*)carve((size_t)wM[l] * wK32[l] * 2);
  float* stats = (float*)carve(32 * 4);
  float* mv    = (float*)carve(32 * 4);

  // ---- convert + pad weights to f16
  const float* wSrc[6] = {preW[0], preW[1], preW[2], postW[0], postW[1], postW[2]};
  for (int l = 0; l < 6; ++l)
    cvt_pad<<<wM[l], wK32[l], 0, stream>>>(wSrc[l], Wh[l], wK[l], wK32[l]);

  // ---- grouping + PPF features
  ball_group_kernel<<<(BATCH * NPTS) / 8, 256, 0, stream>>>(xyz, idxBuf);
  ppf_kernel<<<(unsigned)(NC_PRE / 256), 256, 0, stream>>>(xyz, nrm, idxBuf, ppfh);

  // ---- pre layers: conv(1x1)+GN+ReLU on NC_PRE columns (cbShift=18)
  struct Layer { const _Float16* x; int M, K32; float invCnt;
                 const float* b; const float* g; const float* bt; const _Float16* w; };
  Layer L[3] = {
    {ppfh,  96, 32, 1.f / (12.f * (float)SN), preB[0], preG[0], preBt[0], Wh[0]},
    {RB,    96, 96, 1.f / (12.f * (float)SN), preB[1], preG[1], preBt[1], Wh[1]},
    {RB,   192, 96, 1.f / (24.f * (float)SN), preB[2], preG[2], preBt[2], Wh[2]},
  };
  for (int l = 0; l < 3; ++l) {
    zero_f32<<<1, 32, 0, stream>>>(stats, 32);
    gemm_wmma<<<dim3((unsigned)(NC_PRE / 512), L[l].M / 16), 256, 0, stream>>>(
        L[l].x, L[l].w, L[l].b, RA, stats, L[l].M, L[l].K32, NC_PRE, 18);
    finalize_stats<<<1, 16, 0, stream>>>(stats, mv, L[l].invCnt);
    gn_relu_col<<<(unsigned)(NC_PRE / 256), 256, 0, stream>>>(
        RA, RB, mv, L[l].g, L[l].bt, L[l].M, 18);
  }

  // ---- max over samples -> pooled [bn][192]
  maxpool_col<<<(unsigned)NC_POST, 192, 0, stream>>>(RB, pooled);

  // ---- post layers on NC_POST columns (cbShift=12)
  // post0: 192->192, GN+ReLU
  zero_f32<<<1, 32, 0, stream>>>(stats, 32);
  gemm_wmma<<<dim3((unsigned)(NC_POST / 512), 192 / 16), 256, 0, stream>>>(
      pooled, Wh[3], postB[0], SA, stats, 192, 192, NC_POST, 12);
  finalize_stats<<<1, 16, 0, stream>>>(stats, mv, 1.f / (24.f * 4096.f));
  gn_relu_col<<<(unsigned)(NC_POST / 256), 256, 0, stream>>>(
      SA, SB, mv, postG[0], postBt[0], 192, 12);
  // post1: 192->96, GN+ReLU
  zero_f32<<<1, 32, 0, stream>>>(stats, 32);
  gemm_wmma<<<dim3((unsigned)(NC_POST / 512), 96 / 16), 256, 0, stream>>>(
      SB, Wh[4], postB[1], SA, stats, 96, 192, NC_POST, 12);
  finalize_stats<<<1, 16, 0, stream>>>(stats, mv, 1.f / (12.f * 4096.f));
  gn_relu_col<<<(unsigned)(NC_POST / 256), 256, 0, stream>>>(
      SA, SB, mv, postG[1], postBt[1], 96, 12);
  // post2: 96->96, bias only (no GN)
  gemm_wmma<<<dim3((unsigned)(NC_POST / 512), 96 / 16), 256, 0, stream>>>(
      SB, Wh[5], postB[2], SA, nullptr, 96, 96, NC_POST, 12);

  // ---- L2 normalize -> d_out (B, N, 96) f32
  final_norm<<<(unsigned)(NC_POST / 256), 256, 0, stream>>>(SA, (float*)d_out);
  (void)in_sizes; (void)n_in; (void)out_size; (void)ws_size;
}
